// SLSTM_56521769615654
// MI455X (gfx1250) — matrix operand order repaired and checked by reading, not compiled, not run on hardware
//
#include <hip/hip_runtime.h>
#include <hip/hip_bf16.h>

typedef __bf16 bf16_t;
typedef __attribute__((ext_vector_type(16))) __bf16 v16bf;
typedef __attribute__((ext_vector_type(8)))  float  v8f;

#define IN_DIM   256
#define HID      1024
#define GDIM     4096      // 4*HID
#define BB       256
#define TT       256

// ---------- WMMA fragment packing maps (CDNA5 ISA 7.12.2, wave32) ----------
// 16-bit A-matrix 16x32 (MxK): lanes 0-15 hold M=lane; K groups of 8:
//   group0 (K 0-7):  lane=M,    elem=K%8
//   group1 (K 8-15): lane=M+16, elem=K%8
//   group2 (K16-23): lane=M,    elem=8+K%8
//   group3 (K24-31): lane=M+16, elem=8+K%8
__device__ __forceinline__ int a_pack_idx(int Ml, int kt) {
    int grp  = kt >> 3;
    int lane = Ml + ((grp & 1) << 4);
    int e    = ((grp >> 1) << 3) + (kt & 7);
    return lane * 16 + e;
}
// 16-bit B-matrix 32x16 (KxN): same K grouping as A: lanes 0-15 -> K 0-7,16-23; lanes 16-31 -> K 8-15,24-31
__device__ __forceinline__ int b_pack_idx(int N, int kt) {
    int lane = (((kt >> 3) & 1) << 4) + N;
    int e    = ((kt >> 4) << 3) + (kt & 7);
    return lane * 16 + e;
}

// ---------- fast activations (v_exp_f32 / v_rcp_f32) ----------
__device__ __forceinline__ float fexp(float x) {
    return __builtin_amdgcn_exp2f(x * 1.4426950408889634f);
}
__device__ __forceinline__ float fsigmoid(float x) {
    return __builtin_amdgcn_rcpf(1.0f + fexp(-x));
}
__device__ __forceinline__ float ftanh(float x) {
    float e2 = fexp(2.0f * x);
    return 1.0f - 2.0f * __builtin_amdgcn_rcpf(e2 + 1.0f);
}

// ---------- prep: fold Wy into W row 255, convert to bf16 B-fragments ----------
__global__ void prep_W_k(const float* __restrict__ W, const float* __restrict__ Wy,
                         bf16_t* __restrict__ Wp) {
    int idx = blockIdx.x * blockDim.x + threadIdx.x;     // d*4096 + g
    if (idx >= IN_DIM * GDIM) return;
    int d = idx >> 12, g = idx & (GDIM - 1);
    float v = W[idx];
    if (d == IN_DIM - 1) v += Wy[g];
    int ntile = g >> 4, N = g & 15, ktile = d >> 5, kt = d & 31;
    Wp[((size_t)ntile * 8 + ktile) * 512 + b_pack_idx(N, kt)] = (bf16_t)v;
}

__global__ void prep_U_k(const float* __restrict__ U, bf16_t* __restrict__ Up) {
    int idx = blockIdx.x * blockDim.x + threadIdx.x;     // d*4096 + g, d<1024
    if (idx >= HID * GDIM) return;
    int d = idx >> 12, g = idx & (GDIM - 1);
    int ntile = g >> 4, N = g & 15, ktile = d >> 5, kt = d & 31;
    Up[((size_t)ntile * 32 + ktile) * 512 + b_pack_idx(N, kt)] = (bf16_t)U[idx];
}

__global__ void prep_X_k(const float* __restrict__ X, bf16_t* __restrict__ Xp) {
    int idx = blockIdx.x * blockDim.x + threadIdx.x;     // b*T*256 + t*256 + d
    if (idx >= BB * TT * IN_DIM) return;
    int d = idx & 255, t = (idx >> 8) & 255, b = idx >> 16;
    int mtile = b >> 4, Ml = b & 15, ktile = d >> 5, kt = d & 31;
    Xp[(((size_t)t * 16 + mtile) * 8 + ktile) * 512 + a_pack_idx(Ml, kt)] = (bf16_t)X[idx];
}

__global__ void init_state_k(float* __restrict__ c, bf16_t* __restrict__ hp0) {
    int i = blockIdx.x * blockDim.x + threadIdx.x;
    if (i < BB * HID) { c[i] = 0.0f; hp0[i] = (bf16_t)0.0f; }
}

// ---------- fused LSTM step: gates = X[:,t]@W' + h@U + bias ; gate math ----------
// grid = 128 blocks x 256 thr (8 waves). wave -> (mtile, ntile); all 8 waves of a
// block share mtile, so the A panels (X row-tile + h row-tile) are staged in LDS once.
__global__ void __launch_bounds__(256) lstm_step_k(
    const bf16_t* __restrict__ Xp, const bf16_t* __restrict__ Wp,
    const bf16_t* __restrict__ Up, const float*  __restrict__ bias,
    const bf16_t* __restrict__ hp_in, bf16_t* __restrict__ hp_out,
    float* __restrict__ c, float* __restrict__ hidden, int t)
{
    __shared__ bf16_t ldsA[(8 + 32) * 512];              // 40 KB: X-panel + h-panel

    const int tid  = threadIdx.x;
    const int lane = tid & 31;
    const int wid  = blockIdx.x * 8 + (tid >> 5);
    const int mtile = wid >> 6;                          // 0..15 (shared per block)
    const int ntile = wid & 63;                          // 0..63

    // cooperative global->LDS stage of the shared A panels (16 rows of X_t and h)
    {
        const uint4* srcX = (const uint4*)(Xp + ((size_t)t * 16 + mtile) * 8 * 512);
        const uint4* srcH = (const uint4*)(hp_in + (size_t)mtile * 32 * 512);
        uint4* dst = (uint4*)ldsA;
        for (int i = tid; i < 512;  i += 256) dst[i]       = srcX[i];
        for (int i = tid; i < 2048; i += 256) dst[512 + i] = srcH[i];
    }
    __syncthreads();

    v8f zero = {};
    v8f acc[4];
    acc[0] = zero; acc[1] = zero; acc[2] = zero; acc[3] = zero;

    const bf16_t* ldsX = ldsA;
    const bf16_t* ldsH = ldsA + 8 * 512;

    // K-loop over input contribution: X_t (16x256) @ W' (256x64)  [4 gate tiles]
    for (int k = 0; k < 8; ++k) {
        v16bf a = *(const v16bf*)(ldsX + k * 512 + lane * 16);
#pragma unroll
        for (int g = 0; g < 4; ++g) {
            v16bf b = *(const v16bf*)(Wp + (((size_t)(g * 64 + ntile)) * 8 + k) * 512 + lane * 16);
            acc[g] = __builtin_amdgcn_wmma_f32_16x16x32_bf16(
                false, a, false, b, (short)0, acc[g], false, false);
        }
    }
    // K-loop over recurrence: h (16x1024) @ U (1024x64)  [4 gate tiles]
    for (int k = 0; k < 32; ++k) {
        v16bf a = *(const v16bf*)(ldsH + k * 512 + lane * 16);
        if (k + 1 < 32)
            __builtin_prefetch(Up + (((size_t)ntile) * 32 + k + 1) * 512 + lane * 16, 0, 1);
#pragma unroll
        for (int g = 0; g < 4; ++g) {
            v16bf b = *(const v16bf*)(Up + (((size_t)(g * 64 + ntile)) * 32 + k) * 512 + lane * 16);
            acc[g] = __builtin_amdgcn_wmma_f32_16x16x32_bf16(
                false, a, false, b, (short)0, acc[g], false, false);
        }
    }

    // epilogue: C/D layout -> (row = m0 + r + 8*(lane>=16), col = lane&15)
    const int ncol = (ntile << 4) + (lane & 15);         // hidden index 0..1023
    const float bi = bias[ncol];
    const float bf = bias[HID + ncol];
    const float bg = bias[2 * HID + ncol];
    const float bo = bias[3 * HID + ncol];
    const int m0 = mtile << 4;
    const int rbase = ((lane >> 4) << 3);

#pragma unroll
    for (int r = 0; r < 8; ++r) {
        int row = m0 + rbase + r;                        // batch index
        float ig = fsigmoid(acc[0][r] + bi);
        float fg = fsigmoid(acc[1][r] + bf);
        float gg = ftanh   (acc[2][r] + bg);
        float og = fsigmoid(acc[3][r] + bo);
        size_t ci = (size_t)row * HID + ncol;
        float cn = fg * c[ci] + ig * gg;
        float hn = og * ftanh(cn);
        c[ci] = cn;
        hidden[((size_t)row * TT + t) * HID + ncol] = hn;
        // re-pack h into bf16 A-fragment layout for next step
        int kt = ncol & 31, ktile = ncol >> 5, Ml = row & 15;
        hp_out[((size_t)mtile * 32 + ktile) * 512 + a_pack_idx(Ml, kt)] = (bf16_t)hn;
    }
}

// ---------- finalize: y_pred = h_T @ fc_w^T + fc_b ; export h_t, c_t ----------
__global__ void finalize_k(const float* __restrict__ hidden, const float* __restrict__ cbuf,
                           const float* __restrict__ fc_w, const float* __restrict__ fc_b,
                           float* __restrict__ ypred, float* __restrict__ h_t,
                           float* __restrict__ c_t)
{
    __shared__ float red[256];
    int b = blockIdx.x, tid = threadIdx.x;
    const float* h_last = hidden + ((size_t)b * TT + (TT - 1)) * HID;
    float s = 0.0f;
    for (int j = tid; j < HID; j += 256) {
        float hv = h_last[j];
        s += hv * fc_w[j];
        h_t[(size_t)b * HID + j] = hv;
        c_t[(size_t)b * HID + j] = cbuf[(size_t)b * HID + j];
    }
    red[tid] = s;
    __syncthreads();
    for (int off = 128; off > 0; off >>= 1) {
        if (tid < off) red[tid] += red[tid + off];
        __syncthreads();
    }
    if (tid == 0) ypred[b] = red[0] + fc_b[0];
}

extern "C" void kernel_launch(void* const* d_in, const int* in_sizes, int n_in,
                              void* d_out, int out_size, void* d_ws, size_t ws_size,
                              hipStream_t stream) {
    const float* X    = (const float*)d_in[0];
    const float* W    = (const float*)d_in[1];
    const float* U    = (const float*)d_in[2];
    const float* bias = (const float*)d_in[3];
    const float* Wy   = (const float*)d_in[4];
    const float* fc_w = (const float*)d_in[5];
    const float* fc_b = (const float*)d_in[6];

    char* ws = (char*)d_ws;
    size_t off = 0;
    auto take = [&](size_t bytes) -> void* {
        void* p = ws + off;
        off = (off + bytes + 255) & ~(size_t)255;
        return p;
    };
    bf16_t* Wp  = (bf16_t*)take((size_t)256 * 8 * 512 * 2);          // 2 MB
    bf16_t* Up  = (bf16_t*)take((size_t)256 * 32 * 512 * 2);         // 8 MB
    bf16_t* Xp  = (bf16_t*)take((size_t)TT * 16 * 8 * 512 * 2);      // 32 MB
    bf16_t* hp0 = (bf16_t*)take((size_t)BB * HID * 2);               // 512 KB
    bf16_t* hp1 = (bf16_t*)take((size_t)BB * HID * 2);               // 512 KB
    float*  cbf = (float*) take((size_t)BB * HID * 4);               // 1 MB

    float* out    = (float*)d_out;
    float* ypred  = out;                                             // (B,1)
    float* hidden = out + BB;                                        // (B,T,H)
    float* h_t    = hidden + (size_t)BB * TT * HID;                  // (B,H)
    float* c_t    = h_t + (size_t)BB * HID;                          // (B,H)

    prep_W_k<<<(IN_DIM * GDIM) / 256, 256, 0, stream>>>(W, Wy, Wp);
    prep_U_k<<<(HID * GDIM) / 256, 256, 0, stream>>>(U, Up);
    prep_X_k<<<(BB * TT * IN_DIM) / 256, 256, 0, stream>>>(X, Xp);
    init_state_k<<<(BB * HID) / 256, 256, 0, stream>>>(cbf, hp0);

    for (int t = 0; t < TT; ++t) {
        const bf16_t* hin = (t & 1) ? hp1 : hp0;
        bf16_t*       hout = (t & 1) ? hp0 : hp1;
        lstm_step_k<<<128, 256, 0, stream>>>(Xp, Wp, Up, bias, hin, hout, cbf, hidden, t);
    }

    finalize_k<<<BB, 256, 0, stream>>>(hidden, cbf, fc_w, fc_b, ypred, h_t, c_t);
}
